// WSDDN_57947698757638
// MI455X (gfx1250) — matrix operand order repaired and checked by reading, not compiled, not run on hardware
//
#include <hip/hip_runtime.h>
#include <stdint.h>

// ---- problem constants (from reference) ----
#define B_    16
#define C_    512
#define TG    4096
#define P_    512
#define MROWS (B_*P_)      // 8192
#define KF    3584         // C * (1+2+4)
#define H6    1024
#define H7    512
#define NC    30

typedef unsigned short u16;
typedef __attribute__((ext_vector_type(16))) __bf16 v16bf;
typedef __attribute__((ext_vector_type(8)))  __bf16 v8bf;
typedef __attribute__((ext_vector_type(8)))  float  v8f;
typedef __attribute__((ext_vector_type(4)))  int    v4i;

__device__ __forceinline__ u16 f2bf(float f) {
    unsigned u = __float_as_uint(f);
    return (u16)((u + 0x7FFFu + ((u >> 16) & 1u)) >> 16);   // RNE
}

// -------- CDNA5 async Global->LDS copy (ASYNCcnt) with sync fallback --------
#if defined(__has_builtin)
#  if __has_builtin(__builtin_amdgcn_global_load_async_to_lds_b128)
#    define HAVE_ASYNC_LDS 1
#  endif
#endif
#ifndef HAVE_ASYNC_LDS
#  define HAVE_ASYNC_LDS 0
#endif

typedef __attribute__((address_space(1))) v4i* gv4i_p;   // global (AS1) int4*
typedef __attribute__((address_space(3))) v4i* lv4i_p;   // LDS    (AS3) int4*

__device__ __forceinline__ void cp16(u16* l, const u16* g) {
#if HAVE_ASYNC_LDS
    __builtin_amdgcn_global_load_async_to_lds_b128((gv4i_p)g, (lv4i_p)l, 0, 0);
#else
    *(uint4*)l = *(const uint4*)g;
#endif
}
__device__ __forceinline__ void wait_async() {
#if HAVE_ASYNC_LDS
#  if __has_builtin(__builtin_amdgcn_s_wait_asynccnt)
    __builtin_amdgcn_s_wait_asynccnt(0);
#  else
    asm volatile("s_wait_asynccnt 0" ::: "memory");
#  endif
#endif
}

// ---------------------------------------------------------------
// K1: transpose-with-shift  x[b][c][t] -> csT[b][t+1][c]; zero csT[b][0][:]
// block (32,32), grid (TG/32, C/32, B)
// ---------------------------------------------------------------
__global__ void k_transpose_shift(const float* __restrict__ x, float* __restrict__ csT) {
    __shared__ float tile[32][33];
    const int b  = blockIdx.z;
    const int c0 = blockIdx.y * 32;
    const int t0 = blockIdx.x * 32;
    const int tx = threadIdx.x, ty = threadIdx.y;
    tile[ty][tx] = x[((size_t)b * C_ + (c0 + ty)) * TG + (t0 + tx)];
    __syncthreads();
    csT[((size_t)b * (TG + 1) + (1 + t0 + ty)) * C_ + (c0 + tx)] = tile[tx][ty];
    if (blockIdx.x == 0 && ty == 0)
        csT[((size_t)b * (TG + 1)) * C_ + (c0 + tx)] = 0.f;
}

// ---------------------------------------------------------------
// K2: per-segment sums (8 segments of 512 t's). grid (C/256, 8, B), block 256
// ---------------------------------------------------------------
__global__ void k_segsum(const float* __restrict__ csT, float* __restrict__ seg) {
    const int c = blockIdx.x * 256 + threadIdx.x;
    const int s = blockIdx.y, b = blockIdx.z;
    const float* base = csT + ((size_t)b * (TG + 1) + 1 + (size_t)s * 512) * C_ + c;
    float acc = 0.f;
    for (int i = 0; i < 512; ++i) acc += base[(size_t)i * C_];
    seg[((size_t)b * 8 + s) * C_ + c] = acc;
}

// ---------------------------------------------------------------
// K3: in-place cumsum along t with segment carry. grid (C/256, 8, B), block 256
// ---------------------------------------------------------------
__global__ void k_scan(float* __restrict__ csT, const float* __restrict__ seg) {
    const int c = blockIdx.x * 256 + threadIdx.x;
    const int s = blockIdx.y, b = blockIdx.z;
    float run = 0.f;
    for (int s2 = 0; s2 < s; ++s2) run += seg[((size_t)b * 8 + s2) * C_ + c];
    float* base = csT + ((size_t)b * (TG + 1) + 1 + (size_t)s * 512) * C_ + c;
    for (int i = 0; i < 512; ++i) {
        run += base[(size_t)i * C_];
        base[(size_t)i * C_] = run;
    }
}

// ---------------------------------------------------------------
// K4: SPP gather -> bf16 feats[row][3584]. grid(MROWS), block 256.
// ---------------------------------------------------------------
__global__ void k_spp(const float* __restrict__ csT, const int* __restrict__ boxes,
                      u16* __restrict__ feats) {
    const int row = blockIdx.x;        // b*P + p
    const int b   = row / P_;
    const int s0  = boxes[row * 2 + 0];
    const int e0  = boxes[row * 2 + 1];
    int st = s0 < 0 ? 0 : (s0 > TG - 1 ? TG - 1 : s0);
    int en = e0 < TG ? e0 : TG;
    if (en < st + 1) en = st + 1;
    const int T = en - st;

    int bs[7], be[7];
    bs[0] = st; be[0] = en;
    #pragma unroll
    for (int i = 0; i < 2; ++i) { bs[1 + i] = st + (i * T) / 2; be[1 + i] = st + ((i + 1) * T + 1) / 2; }
    #pragma unroll
    for (int i = 0; i < 4; ++i) { bs[3 + i] = st + (i * T) / 4; be[3 + i] = st + ((i + 1) * T + 3) / 4; }

    const float* crow = csT + (size_t)b * (TG + 1) * C_;
    u16* frow = feats + (size_t)row * KF;
    for (int c = threadIdx.x; c < C_; c += blockDim.x) {
        float v[7];
        #pragma unroll
        for (int j = 0; j < 7; ++j) {
            float d = crow[(size_t)be[j] * C_ + c] - crow[(size_t)bs[j] * C_ + c];
            v[j] = d / (float)(be[j] - bs[j]);
        }
        frow[c]                = f2bf(v[0]);
        frow[512  + 2 * c + 0] = f2bf(v[1]);
        frow[512  + 2 * c + 1] = f2bf(v[2]);
        frow[1536 + 4 * c + 0] = f2bf(v[3]);
        frow[1536 + 4 * c + 1] = f2bf(v[4]);
        frow[1536 + 4 * c + 2] = f2bf(v[5]);
        frow[1536 + 4 * c + 3] = f2bf(v[6]);
    }
}

// ---------------------------------------------------------------
// K5: f32 -> bf16 convert; K5b: pack [Wc;Wd;0] into 64x512 bf16
// ---------------------------------------------------------------
__global__ void k_f2bf(const float* __restrict__ src, u16* __restrict__ dst, int n) {
    int i = blockIdx.x * 256 + threadIdx.x;
    if (i < n) dst[i] = f2bf(src[i]);
}
__global__ void k_pack_heads(const float* __restrict__ Wc, const float* __restrict__ Wd,
                             u16* __restrict__ dst) {
    int i = blockIdx.x * 256 + threadIdx.x;   // 64*512
    int n = i >> 9, k = i & 511;
    float v = 0.f;
    if (n < NC)          v = Wc[n * 512 + k];
    else if (n < 2 * NC) v = Wd[(n - NC) * 512 + k];
    dst[i] = f2bf(v);
}

// ---------------------------------------------------------------
// LDS-staged WMMA GEMM:  out[M][N] = relu(A[M][K] * Bw[N][K]^T + bias)
// block 256 = 8 waves; block tile 128M x 128N; wave tile 32M x 64N (8 WMMAs/step)
// K-step 32, double-buffered LDS filled with async Global->LDS b128 copies.
// grid (N/128, M/128).
// ---------------------------------------------------------------
#define LSTR 48   // padded LDS row stride in u16 elems (96B: 32B-aligned, low conflicts)

__global__ void k_gemm_relu(const u16* __restrict__ A, const u16* __restrict__ Bw,
                            const float* __restrict__ bias,
                            u16* __restrict__ outB, float* __restrict__ outF,
                            int M, int N, int K) {
    __shared__ u16 ldsA[2][128 * LSTR];
    __shared__ u16 ldsB[2][128 * LSTR];

    const int tid   = threadIdx.x;
    const int wave  = tid >> 5;
    const int lane  = tid & 31;
    const int half  = lane >> 4;
    const int l16   = lane & 15;
    const int mW    = (wave & 3) * 32;   // wave M offset in block tile
    const int nW    = (wave >> 2) * 64;  // wave N offset in block tile
    const int mBlk  = blockIdx.y * 128;
    const int nBlk  = blockIdx.x * 128;

    const u16* Ablk = A  + (size_t)mBlk * K;
    const u16* Bblk = Bw + (size_t)nBlk * K;

    // copy-chunk mapping: 128 rows x 4 x 16B chunks = 512 chunks; 2 per thread
    const int q0r = tid >> 2,          q0s = tid & 3;
    const int q1r = (tid + 256) >> 2,  q1s = (tid + 256) & 3;

    v8f acc[2][4] = {};
    const int nk = K >> 5;

    // prologue: fill buffer 0
    {
        cp16(&ldsA[0][q0r * LSTR + q0s * 8], Ablk + (size_t)q0r * K + q0s * 8);
        cp16(&ldsA[0][q1r * LSTR + q1s * 8], Ablk + (size_t)q1r * K + q1s * 8);
        cp16(&ldsB[0][q0r * LSTR + q0s * 8], Bblk + (size_t)q0r * K + q0s * 8);
        cp16(&ldsB[0][q1r * LSTR + q1s * 8], Bblk + (size_t)q1r * K + q1s * 8);
    }

    for (int i = 0; i < nk; ++i) {
        const int cur = i & 1;
        wait_async();
        __syncthreads();                       // buf[cur] ready for all waves
        if (i + 1 < nk) {                      // prefetch next K-slab into buf[1-cur]
            const int k1 = (i + 1) << 5;
            cp16(&ldsA[1 - cur][q0r * LSTR + q0s * 8], Ablk + (size_t)q0r * K + k1 + q0s * 8);
            cp16(&ldsA[1 - cur][q1r * LSTR + q1s * 8], Ablk + (size_t)q1r * K + k1 + q1s * 8);
            cp16(&ldsB[1 - cur][q0r * LSTR + q0s * 8], Bblk + (size_t)q0r * K + k1 + q0s * 8);
            cp16(&ldsB[1 - cur][q1r * LSTR + q1s * 8], Bblk + (size_t)q1r * K + k1 + q1s * 8);
        }

        // A fragments (2 M-tiles), contiguous k = 16*half + s per lane half (same order as B)
        v16bf afrag[2];
        #pragma unroll
        for (int ma = 0; ma < 2; ++ma) {
            const u16* ar = &ldsA[cur][(mW + ma * 16 + l16) * LSTR + half * 16];
            v8bf lo = *(const v8bf*)ar;
            v8bf hi = *(const v8bf*)(ar + 8);
            afrag[ma] = __builtin_shufflevector(lo, hi, 0,1,2,3,4,5,6,7,8,9,10,11,12,13,14,15);
        }
        #pragma unroll
        for (int t = 0; t < 4; ++t) {
            const u16* br = &ldsB[cur][(nW + t * 16 + l16) * LSTR + half * 16];
            v16bf bf = *(const v16bf*)br;       // 32B-aligned (LSTR*2 % 32 == 0)
            acc[0][t] = __builtin_amdgcn_wmma_f32_16x16x32_bf16(
                false, afrag[0], false, bf, (short)0, acc[0][t], false, false);
            acc[1][t] = __builtin_amdgcn_wmma_f32_16x16x32_bf16(
                false, afrag[1], false, bf, (short)0, acc[1][t], false, false);
        }
        __syncthreads();                       // all reads of buf[cur] done
    }

    #pragma unroll
    for (int ma = 0; ma < 2; ++ma) {
        #pragma unroll
        for (int t = 0; t < 4; ++t) {
            const int n = nBlk + nW + t * 16 + l16;
            const float bv = bias[n];
            #pragma unroll
            for (int r = 0; r < 8; ++r) {
                const int m = mBlk + mW + ma * 16 + half * 8 + r;
                float v = acc[ma][t][r] + bv;
                v = v > 0.f ? v : 0.f;
                outB[(size_t)m * N + n] = f2bf(v);
                if (outF) outF[(size_t)m * N + n] = v;
            }
        }
    }
}

// ---------------------------------------------------------------
// Heads GEMM: h7b[8192][512] x Wcd[64][512]^T -> class/det logits (direct loads)
// ---------------------------------------------------------------
__global__ void k_heads(const u16* __restrict__ A, const u16* __restrict__ Bw,
                        const float* __restrict__ bc, const float* __restrict__ bd,
                        float* __restrict__ clsOut, float* __restrict__ detOut) {
    const int K = H7, tid = threadIdx.x;
    const int wave = tid >> 5, lane = tid & 31;
    const int half = lane >> 4, l16 = lane & 15;
    const int m0 = blockIdx.x * 128 + wave * 16;

    v8f acc[4] = {};
    const u16* aBase = A  + (size_t)(m0 + l16) * K + half * 16;
    const u16* bBase = Bw + (size_t)l16 * K + half * 16;

    for (int k0 = 0; k0 < K; k0 += 32) {
        v8bf alo = *(const v8bf*)(aBase + k0);
        v8bf ahi = *(const v8bf*)(aBase + k0 + 8);
        v16bf a = __builtin_shufflevector(alo, ahi, 0,1,2,3,4,5,6,7,8,9,10,11,12,13,14,15);
        #pragma unroll
        for (int t = 0; t < 4; ++t) {
            v16bf bf = *(const v16bf*)(bBase + (size_t)t * 16 * K + k0);
            acc[t] = __builtin_amdgcn_wmma_f32_16x16x32_bf16(
                false, a, false, bf, (short)0, acc[t], false, false);
        }
    }

    #pragma unroll
    for (int t = 0; t < 4; ++t) {
        const int n = t * 16 + l16;
        #pragma unroll
        for (int r = 0; r < 8; ++r) {
            const int m = m0 + half * 8 + r;
            const float v = acc[t][r];
            if (n < NC)          clsOut[(size_t)m * NC + n]        = v + bc[n];
            else if (n < 2 * NC) detOut[(size_t)m * NC + (n - NC)] = v + bd[n - NC];
        }
    }
}

// ---------------------------------------------------------------
// K8: class softmax over 30 classes -> write probs into joint slot
// ---------------------------------------------------------------
__global__ void k_clssm(const float* __restrict__ logits, float* __restrict__ prob) {
    const int m = blockIdx.x * 256 + threadIdx.x;
    if (m >= MROWS) return;
    const float* L = logits + (size_t)m * NC;
    float mx = L[0];
    #pragma unroll
    for (int j = 1; j < NC; ++j) mx = fmaxf(mx, L[j]);
    float e[NC], s = 0.f;
    #pragma unroll
    for (int j = 0; j < NC; ++j) { e[j] = __expf(L[j] - mx); s += e[j]; }
    const float inv = 1.f / s;
    #pragma unroll
    for (int j = 0; j < NC; ++j) prob[(size_t)m * NC + j] = e[j] * inv;
}

// ---------------------------------------------------------------
// K9: det softmax over 512 proposals. grid(B*NC), block 256 (2 p's per thread)
// ---------------------------------------------------------------
__global__ void k_detsm(const float* __restrict__ logits, float* __restrict__ dprob) {
    __shared__ float red[256];
    const int b = blockIdx.x / NC, cls = blockIdx.x % NC;
    const int tid = threadIdx.x;
    const size_t base = ((size_t)b * P_) * NC + cls;
    const float v0 = logits[base + (size_t)tid * NC];
    const float v1 = logits[base + (size_t)(tid + 256) * NC];
    red[tid] = fmaxf(v0, v1);
    __syncthreads();
    for (int o = 128; o > 0; o >>= 1) { if (tid < o) red[tid] = fmaxf(red[tid], red[tid + o]); __syncthreads(); }
    const float mx = red[0];
    __syncthreads();
    const float e0 = __expf(v0 - mx), e1 = __expf(v1 - mx);
    red[tid] = e0 + e1;
    __syncthreads();
    for (int o = 128; o > 0; o >>= 1) { if (tid < o) red[tid] += red[tid + o]; __syncthreads(); }
    const float inv = 1.f / red[0];
    dprob[base + (size_t)tid * NC] = e0 * inv;
    dprob[base + (size_t)(tid + 256) * NC] = e1 * inv;
}

// ---------------------------------------------------------------
// K10: joint = class_prob * det_prob (in place on joint slot) + video sum over p
// ---------------------------------------------------------------
__global__ void k_joint(const float* __restrict__ dprob, float* __restrict__ joint,
                        float* __restrict__ video) {
    __shared__ float red[256];
    const int b = blockIdx.x / NC, cls = blockIdx.x % NC;
    const int tid = threadIdx.x;
    const size_t base = ((size_t)b * P_) * NC + cls;
    const float j0 = joint[base + (size_t)tid * NC] * dprob[base + (size_t)tid * NC];
    const float j1 = joint[base + (size_t)(tid + 256) * NC] * dprob[base + (size_t)(tid + 256) * NC];
    joint[base + (size_t)tid * NC] = j0;
    joint[base + (size_t)(tid + 256) * NC] = j1;
    red[tid] = j0 + j1;
    __syncthreads();
    for (int o = 128; o > 0; o >>= 1) { if (tid < o) red[tid] += red[tid + o]; __syncthreads(); }
    if (tid == 0) video[b * NC + cls] = red[0];
}

// ---------------------------------------------------------------
extern "C" void kernel_launch(void* const* d_in, const int* in_sizes, int n_in,
                              void* d_out, int out_size, void* d_ws, size_t ws_size,
                              hipStream_t stream) {
    const float* x     = (const float*)d_in[0];
    const int*   boxes = (const int*)  d_in[1];
    const float* W6    = (const float*)d_in[2];
    const float* b6    = (const float*)d_in[3];
    const float* W7    = (const float*)d_in[4];
    const float* b7    = (const float*)d_in[5];
    const float* Wc    = (const float*)d_in[6];
    const float* bc    = (const float*)d_in[7];
    const float* Wd    = (const float*)d_in[8];
    const float* bd    = (const float*)d_in[9];

    float* out = (float*)d_out;
    float* video  = out;                                      // [16,30]
    float* joint  = out + 480;                                // [16,512,30]
    float* clslog = out + 480 + 245760;                       // [16,512,30]
    float* detlog = out + 480 + 2 * 245760;                   // [16,512,30]
    float* h7out  = out + 480 + 3 * 245760;                   // [16,512,512]

    char* ws = (char*)d_ws;
    size_t off = 0;
    float* csT  = (float*)(ws + off); off += (size_t)B_ * (TG + 1) * C_ * 4;   // 134.25 MB
    float* seg  = (float*)(ws + off); off += (size_t)B_ * 8 * C_ * 4;          // 256 KB
    u16* feats  = (u16*)  (ws + off); off += (size_t)MROWS * KF * 2;           // 58.7 MB
    u16* W6b    = (u16*)  (ws + off); off += (size_t)H6 * KF * 2;              // 7.3 MB
    u16* W7b    = (u16*)  (ws + off); off += (size_t)H7 * H6 * 2;              // 1 MB
    u16* Wcdb   = (u16*)  (ws + off); off += (size_t)64 * H7 * 2;              // 64 KB
    u16* h6b    = (u16*)  (ws + off); off += (size_t)MROWS * H6 * 2;           // 16.8 MB
    u16* h7b    = (u16*)  (ws + off); off += (size_t)MROWS * H7 * 2;           // 8.4 MB
    float* dprob= (float*)(ws + off); off += (size_t)MROWS * NC * 4;           // 983 KB

    k_transpose_shift<<<dim3(TG / 32, C_ / 32, B_), dim3(32, 32), 0, stream>>>(x, csT);
    k_segsum<<<dim3(C_ / 256, 8, B_), 256, 0, stream>>>(csT, seg);
    k_scan  <<<dim3(C_ / 256, 8, B_), 256, 0, stream>>>(csT, seg);
    k_spp<<<MROWS, 256, 0, stream>>>(csT, boxes, feats);
    k_f2bf<<<(H6 * KF + 255) / 256, 256, 0, stream>>>(W6, W6b, H6 * KF);
    k_f2bf<<<(H7 * H6 + 255) / 256, 256, 0, stream>>>(W7, W7b, H7 * H6);
    k_pack_heads<<<(64 * H7) / 256, 256, 0, stream>>>(Wc, Wd, Wcdb);
    // FC6: 8192x3584x1024
    k_gemm_relu<<<dim3(H6 / 128, MROWS / 128), 256, 0, stream>>>(
        feats, W6b, b6, h6b, (float*)nullptr, MROWS, H6, KF);
    // FC7: 8192x1024x512 (+f32 h7 to output)
    k_gemm_relu<<<dim3(H7 / 128, MROWS / 128), 256, 0, stream>>>(
        h6b, W7b, b7, h7b, h7out, MROWS, H7, H6);
    k_heads<<<MROWS / 128, 256, 0, stream>>>(h7b, Wcdb, bc, bd, clslog, detlog);
    k_clssm<<<MROWS / 256, 256, 0, stream>>>(clslog, joint);
    k_detsm<<<B_ * NC, 256, 0, stream>>>(detlog, dprob);
    k_joint<<<B_ * NC, 256, 0, stream>>>(dprob, joint, video);

    (void)in_sizes; (void)n_in; (void)out_size; (void)ws_size;
}
